// GraphormerModel_78391743086868
// MI455X (gfx1250) — matrix operand order repaired and checked by reading, not compiled, not run on hardware
//
#include <hip/hip_runtime.h>
#include <hip/hip_bf16.h>

// ---------------------------------------------------------------------------
// Graphormer encoder layer for MI455X (gfx1250), wave32 + bf16 WMMA.
// All GEMMs are NT (row-major A [M,K], row-major B [N,K]) which maps directly
// onto the CDNA5 bf16 WMMA fragment layouts with contiguous b128 loads.
// Attention is computed as S^T = K*Q^T, O^T = V^T * P^T with online softmax,
// so the [8,3072,3072] score tensor is never materialized.
// ---------------------------------------------------------------------------

typedef __bf16 bf16_t;
typedef __attribute__((ext_vector_type(8)))  bf16_t v8bf;
typedef __attribute__((ext_vector_type(16))) bf16_t v16bf;
typedef __attribute__((ext_vector_type(8)))  float  v8f;

union F8  { v8f v; float f[8]; };
union B16 { v16bf v; unsigned u[8]; };

#define SHUF16(lo, hi) __builtin_shufflevector(lo, hi, 0,1,2,3,4,5,6,7,8,9,10,11,12,13,14,15)

static __device__ __forceinline__ unsigned pack2bf(float a, float b) {
  bf16_t x = (bf16_t)a, y = (bf16_t)b;
  unsigned short ux, uy;
  __builtin_memcpy(&ux, &x, 2);
  __builtin_memcpy(&uy, &y, 2);
  return (unsigned)ux | ((unsigned)uy << 16);
}

// ---------------------------------------------------------------------------
// Generic NT GEMM: out[m,n] = sum_k A[m,k]*B[n,k]  (+bias)*scale, relu,
// +residual, stored as f32 and/or bf16. One wave computes a 16x64 tile.
// A-fragment (16x32 bf16): lane m=L&15, halves i -> K=(L>=16?16:0)+i (contig).
// B-fragment (32x16 bf16): lane n=L&15, halves i -> K=(L>=16?16:0)+i (contig).
// ---------------------------------------------------------------------------
__global__ __launch_bounds__(256) void gemm_nt(
    const bf16_t* __restrict__ A, const bf16_t* __restrict__ B,
    const float* __restrict__ bias, const float* __restrict__ residual,
    float* __restrict__ outF, bf16_t* __restrict__ outB,
    int M, int N, int K, float scale, int do_relu)
{
  const int wid  = (blockIdx.x << 3) + (threadIdx.x >> 5);
  const int lane = threadIdx.x & 31;
  const int hi   = lane >> 4;      // half-wave select
  const int l15  = lane & 15;
  const int ntiles = N >> 6;
  const int m0 = (wid / ntiles) << 4;
  const int n0 = (wid % ntiles) << 6;

  F8 acc[4] = {};
  const bf16_t* arow = A + (size_t)(m0 + l15) * K + (hi ? 16 : 0);
  const size_t  boff = (size_t)(n0 + l15) * K + (hi ? 16 : 0);

  for (int k0 = 0; k0 < K; k0 += 32) {
    v8bf alo = *(const v8bf*)(arow + k0);
    v8bf ahi = *(const v8bf*)(arow + k0 + 8);
    v16bf af = SHUF16(alo, ahi);
#pragma unroll
    for (int t = 0; t < 4; ++t) {
      v16bf bf = *(const v16bf*)(B + boff + (size_t)(t * 16) * K + k0);
      acc[t].v = __builtin_amdgcn_wmma_f32_16x16x32_bf16(
          false, af, false, bf, (short)0, acc[t].v, false, false);
    }
  }

#pragma unroll
  for (int t = 0; t < 4; ++t) {
    const int col = n0 + t * 16 + l15;
    const float bv = bias ? bias[col] : 0.0f;
#pragma unroll
    for (int r = 0; r < 8; ++r) {
      const int row = m0 + r + (hi << 3);
      float v = (acc[t].f[r] + bv) * scale;
      if (do_relu) v = fmaxf(v, 0.0f);
      const size_t o = (size_t)row * N + col;
      if (residual) v += residual[o];
      if (outF) outF[o] = v;
      if (outB) outB[o] = (bf16_t)v;
    }
  }
}

// ---------------------------------------------------------------------------
// Flash attention, one wave per (head, 16-query block). Q pre-scaled by 1/8.
//   S^T(16k x 16q) = K_tile(A) * Q^T(B)  [2 dim-chunks of 32]
//   bias from spatial[dist[q,k], h] gathered per tile
//   online softmax per query column (lane pair {L, L+16})
//   O^T(16d x 16q) += V^T_tile(A) * P^T(B); P^T built via shfl_xor(16) swap
// ---------------------------------------------------------------------------
__global__ __launch_bounds__(256) void attn_flash(
    const bf16_t* __restrict__ qb, const bf16_t* __restrict__ kb,
    const bf16_t* __restrict__ vT, const int* __restrict__ dist,
    const float* __restrict__ spatial, bf16_t* __restrict__ outB)
{
  const int NT = 3072;
  const int wid  = (blockIdx.x << 3) + (threadIdx.x >> 5);
  const int lane = threadIdx.x & 31;
  const int hi   = lane >> 4;
  const int l15  = lane & 15;
  const int head = wid / (NT / 16);
  const int i0   = (wid % (NT / 16)) << 4;

  // Q^T B-fragments (dims 0..31 and 32..63), contiguous per lane
  B16 bq0, bq1;
  {
    const bf16_t* base = qb + (size_t)(i0 + l15) * 512 + head * 64 + (hi ? 16 : 0);
    bq0.v = *(const v16bf*)base;
    bq1.v = *(const v16bf*)(base + 32);
  }

  F8 oacc[4] = {};
  float m_run = -3.0e38f, l_run = 0.0f;
  const int* drow = dist + (size_t)(i0 + l15) * NT;

  for (int j = 0; j < NT; j += 32) {
    F8 st[2];
#pragma unroll
    for (int p = 0; p < 2; ++p) {
      const bf16_t* krow = kb + (size_t)(j + p * 16 + l15) * 512 + head * 64 + (hi ? 16 : 0);
      v8bf a0 = *(const v8bf*)(krow);
      v8bf a1 = *(const v8bf*)(krow + 8);
      v8bf a2 = *(const v8bf*)(krow + 32);
      v8bf a3 = *(const v8bf*)(krow + 40);
      v16bf af0 = SHUF16(a0, a1);
      v16bf af1 = SHUF16(a2, a3);
      v8f z = {0.f, 0.f, 0.f, 0.f, 0.f, 0.f, 0.f, 0.f};
      st[p].v = __builtin_amdgcn_wmma_f32_16x16x32_bf16(
          false, af0, false, bq0.v, (short)0, z, false, false);
      st[p].v = __builtin_amdgcn_wmma_f32_16x16x32_bf16(
          false, af1, false, bq1.v, (short)0, st[p].v, false, false);
      // spatial bias: key = j + 16p + 8*hi + r (contiguous), query = i0+l15
      int di[8];
      *(int4*)&di[0] = *(const int4*)(drow + j + p * 16 + (hi << 3));
      *(int4*)&di[4] = *(const int4*)(drow + j + p * 16 + (hi << 3) + 4);
#pragma unroll
      for (int r = 0; r < 8; ++r) st[p].f[r] += spatial[di[r] * 8 + head];
    }

    // online softmax over the 32 keys of this step (per query column)
    float mx = -3.0e38f;
#pragma unroll
    for (int p = 0; p < 2; ++p)
#pragma unroll
      for (int r = 0; r < 8; ++r) mx = fmaxf(mx, st[p].f[r]);
    mx = fmaxf(mx, __shfl_xor(mx, 16));
    const float m_new = fmaxf(m_run, mx);
    const float alpha = __expf(m_run - m_new);
    float psum = 0.0f;
#pragma unroll
    for (int p = 0; p < 2; ++p)
#pragma unroll
      for (int r = 0; r < 8; ++r) {
        float e = __expf(st[p].f[r] - m_new);
        st[p].f[r] = e;
        psum += e;
      }
    psum += __shfl_xor(psum, 16);
    l_run = l_run * alpha + psum;
    m_run = m_new;
#pragma unroll
    for (int t = 0; t < 4; ++t)
#pragma unroll
      for (int r = 0; r < 8; ++r) oacc[t].f[r] *= alpha;

    // pack P^T into the B-fragment via half-wave swap
    unsigned pt0[4], pt1[4];
#pragma unroll
    for (int g = 0; g < 4; ++g) {
      pt0[g] = pack2bf(st[0].f[2 * g], st[0].f[2 * g + 1]);
      pt1[g] = pack2bf(st[1].f[2 * g], st[1].f[2 * g + 1]);
    }
    B16 bp;
#pragma unroll
    for (int g = 0; g < 4; ++g) {
      unsigned s0 = (unsigned)__shfl_xor((int)pt0[g], 16);
      unsigned s1 = (unsigned)__shfl_xor((int)pt1[g], 16);
      bp.u[g]     = hi ? s1 : pt0[g];
      bp.u[4 + g] = hi ? pt1[g] : s0;
    }

    // O^T += V^T * P^T, 4 dim-tiles of 16
#pragma unroll
    for (int t = 0; t < 4; ++t) {
      const bf16_t* vrow = vT + (size_t)(head * 64 + t * 16 + l15) * NT + j + (hi ? 16 : 0);
      v8bf v0 = *(const v8bf*)(vrow);
      v8bf v1 = *(const v8bf*)(vrow + 8);
      v16bf av = SHUF16(v0, v1);
      oacc[t].v = __builtin_amdgcn_wmma_f32_16x16x32_bf16(
          false, av, false, bp.v, (short)0, oacc[t].v, false, false);
    }
  }

  const float inv = 1.0f / l_run;
#pragma unroll
  for (int t = 0; t < 4; ++t)
#pragma unroll
    for (int r = 0; r < 8; ++r)
      outB[(size_t)(i0 + l15) * 512 + head * 64 + t * 16 + r + (hi << 3)] =
          (bf16_t)(oacc[t].f[r] * inv);
}

// ---------------------------------------------------------------------------
// LayerNorm over rows of 512, one wave per row.
// ---------------------------------------------------------------------------
__global__ __launch_bounds__(256) void layernorm512(
    const float* __restrict__ in, const float* __restrict__ g,
    const float* __restrict__ b, float* __restrict__ outF,
    bf16_t* __restrict__ outB, int rows)
{
  const int row  = (blockIdx.x << 3) + (threadIdx.x >> 5);
  const int lane = threadIdx.x & 31;
  if (row >= rows) return;
  const float* x = in + (size_t)row * 512;
  float v[16];
  float s = 0.0f;
#pragma unroll
  for (int i = 0; i < 16; ++i) { v[i] = x[i * 32 + lane]; s += v[i]; }
#pragma unroll
  for (int m = 16; m >= 1; m >>= 1) s += __shfl_xor(s, m);
  const float mu = s * (1.0f / 512.0f);
  float s2 = 0.0f;
#pragma unroll
  for (int i = 0; i < 16; ++i) { float d = v[i] - mu; s2 += d * d; }
#pragma unroll
  for (int m = 16; m >= 1; m >>= 1) s2 += __shfl_xor(s2, m);
  const float rs = rsqrtf(s2 * (1.0f / 512.0f) + 1e-5f);
#pragma unroll
  for (int i = 0; i < 16; ++i) {
    const int c = i * 32 + lane;
    const float y = (v[i] - mu) * rs * g[c] + b[c];
    if (outF) outF[(size_t)row * 512 + c] = y;
    if (outB) outB[(size_t)row * 512 + c] = (bf16_t)y;
  }
}

// ---------------------------------------------------------------------------
// Elementwise helpers
// ---------------------------------------------------------------------------
__global__ void f32_to_bf16_k(const float* __restrict__ in, bf16_t* __restrict__ out, int n) {
  int i = blockIdx.x * 256 + threadIdx.x;
  if (i < n) out[i] = (bf16_t)in[i];
}

// x += deg1[in_deg[row]] + deg2[out_deg[row]]; also emit bf16 copy. 3072x512.
__global__ void deg_add_k(float* __restrict__ x, bf16_t* __restrict__ xb,
                          const int* __restrict__ in_deg, const int* __restrict__ out_deg,
                          const float* __restrict__ deg1, const float* __restrict__ deg2) {
  int i = blockIdx.x * 256 + threadIdx.x;     // exact grid
  int row = i >> 9, c = i & 511;
  float v = x[i] + deg1[in_deg[row] * 512 + c] + deg2[out_deg[row] * 512 + c];
  x[i] = v;
  xb[i] = (bf16_t)v;
}

// v [3072][512] bf16 -> vT [512][3072] bf16 (i.e. [H][64][N])
__global__ void transpose_v_k(const bf16_t* __restrict__ v, bf16_t* __restrict__ vT) {
  int i = blockIdx.x * 256 + threadIdx.x;     // exact grid, 3072*512
  int q = i >> 9, hd = i & 511;
  vT[(size_t)hd * 3072 + q] = v[i];
}

// ---------------------------------------------------------------------------
// Host launcher
// ---------------------------------------------------------------------------
static inline size_t alignup256(size_t x) { return (x + 255) & ~(size_t)255; }

extern "C" void kernel_launch(void* const* d_in, const int* in_sizes, int n_in,
                              void* d_out, int out_size, void* d_ws, size_t ws_size,
                              hipStream_t stream) {
  const int NM = 2048, ND = 1024, NN = 3072, HID = 512;
  const int NR = 2048, NDIS = 2048, FFN = 2048;

  const float* M_feat = (const float*)d_in[0];
  const float* D_feat = (const float*)d_in[1];
  const int*   in_deg = (const int*)d_in[2];
  const int*   out_deg= (const int*)d_in[3];
  const int*   dist   = (const int*)d_in[4];
  const float* fc1_w  = (const float*)d_in[5];
  const float* fc1_b  = (const float*)d_in[6];
  const float* fc2_w  = (const float*)d_in[7];
  const float* fc2_b  = (const float*)d_in[8];
  const float* deg1   = (const float*)d_in[9];
  const float* deg2   = (const float*)d_in[10];
  const float* spatial= (const float*)d_in[11];
  const float* q_w = (const float*)d_in[12]; const float* q_b = (const float*)d_in[13];
  const float* k_w = (const float*)d_in[14]; const float* k_b = (const float*)d_in[15];
  const float* v_w = (const float*)d_in[16]; const float* v_b = (const float*)d_in[17];
  const float* o_w = (const float*)d_in[18]; const float* o_b = (const float*)d_in[19];
  const float* ffn_w1 = (const float*)d_in[20]; const float* ffn_b1 = (const float*)d_in[21];
  const float* ffn_w2 = (const float*)d_in[22]; const float* ffn_b2 = (const float*)d_in[23];
  const float* ln1_g = (const float*)d_in[24]; const float* ln1_b = (const float*)d_in[25];
  const float* ln2_g = (const float*)d_in[26]; const float* ln2_b = (const float*)d_in[27];

  char* ws = (char*)d_ws;
  size_t off = 0;
  auto alloc = [&](size_t bytes) -> char* {
    char* p = ws + off;
    off = alignup256(off + bytes);
    return p;
  };

  bf16_t* mb   = (bf16_t*)alloc((size_t)NM * NR * 2);
  bf16_t* db   = (bf16_t*)alloc((size_t)ND * NDIS * 2);
  bf16_t* w1b  = (bf16_t*)alloc((size_t)HID * NR * 2);
  bf16_t* w2b  = (bf16_t*)alloc((size_t)HID * NDIS * 2);
  bf16_t* qwb  = (bf16_t*)alloc((size_t)HID * HID * 2);
  bf16_t* kwb  = (bf16_t*)alloc((size_t)HID * HID * 2);
  bf16_t* vwb  = (bf16_t*)alloc((size_t)HID * HID * 2);
  bf16_t* owb  = (bf16_t*)alloc((size_t)HID * HID * 2);
  bf16_t* f1wb = (bf16_t*)alloc((size_t)FFN * HID * 2);
  bf16_t* f2wb = (bf16_t*)alloc((size_t)HID * FFN * 2);
  float*  xf   = (float*)alloc((size_t)NN * HID * 4);
  bf16_t* xb   = (bf16_t*)alloc((size_t)NN * HID * 2);
  bf16_t* qbuf = (bf16_t*)alloc((size_t)NN * HID * 2);
  bf16_t* kbuf = (bf16_t*)alloc((size_t)NN * HID * 2);
  bf16_t* vbuf = (bf16_t*)alloc((size_t)NN * HID * 2);
  bf16_t* vTb  = (bf16_t*)alloc((size_t)HID * NN * 2);
  bf16_t* attb = (bf16_t*)alloc((size_t)NN * HID * 2);
  float*  yf   = (float*)alloc((size_t)NN * HID * 4);
  float*  hf   = (float*)alloc((size_t)NN * HID * 4);
  bf16_t* hb   = (bf16_t*)alloc((size_t)NN * HID * 2);
  bf16_t* midb = (bf16_t*)alloc((size_t)NN * FFN * 2);
  float*  y2f  = (float*)alloc((size_t)NN * HID * 4);
  (void)ws_size; (void)n_in; (void)in_sizes; (void)out_size;

  auto cvt = [&](const float* src, bf16_t* dst, int n) {
    f32_to_bf16_k<<<(n + 255) / 256, 256, 0, stream>>>(src, dst, n);
  };
  cvt(M_feat, mb, NM * NR);
  cvt(D_feat, db, ND * NDIS);
  cvt(fc1_w, w1b, HID * NR);
  cvt(fc2_w, w2b, HID * NDIS);
  cvt(q_w, qwb, HID * HID);
  cvt(k_w, kwb, HID * HID);
  cvt(v_w, vwb, HID * HID);
  cvt(o_w, owb, HID * HID);
  cvt(ffn_w1, f1wb, FFN * HID);
  cvt(ffn_w2, f2wb, HID * FFN);

  auto gemm = [&](const bf16_t* A, const bf16_t* B, const float* bias,
                  const float* res, float* oF, bf16_t* oB,
                  int M, int N, int K, float scale, int relu) {
    int blocks = (M / 16) * (N / 64) / 8;
    gemm_nt<<<blocks, 256, 0, stream>>>(A, B, bias, res, oF, oB, M, N, K, scale, relu);
  };

  // x = concat(M@fc1.T + b1, D@fc2.T + b2)
  gemm(mb, w1b, fc1_b, nullptr, xf, nullptr, NM, HID, NR, 1.0f, 0);
  gemm(db, w2b, fc2_b, nullptr, xf + (size_t)NM * HID, nullptr, ND, HID, NDIS, 1.0f, 0);
  // + degree embeddings, emit bf16 copy of x
  deg_add_k<<<(NN * HID) / 256, 256, 0, stream>>>(xf, xb, in_deg, out_deg, deg1, deg2);

  // Q (pre-scaled by HD^-0.5 = 0.125), K, V
  gemm(xb, qwb, q_b, nullptr, nullptr, qbuf, NN, HID, HID, 0.125f, 0);
  gemm(xb, kwb, k_b, nullptr, nullptr, kbuf, NN, HID, HID, 1.0f, 0);
  gemm(xb, vwb, v_b, nullptr, nullptr, vbuf, NN, HID, HID, 1.0f, 0);
  transpose_v_k<<<(NN * HID) / 256, 256, 0, stream>>>(vbuf, vTb);

  // fused biased attention: 8 heads x 192 query blocks = 1536 waves
  attn_flash<<<192, 256, 0, stream>>>(qbuf, kbuf, vTb, dist, spatial, attb);

  // O projection + residual, LN1
  gemm(attb, owb, o_b, xf, yf, nullptr, NN, HID, HID, 1.0f, 0);
  layernorm512<<<NN / 8, 256, 0, stream>>>(yf, ln1_g, ln1_b, hf, hb, NN);

  // FFN + residual, LN2 -> d_out
  gemm(hb, f1wb, ffn_b1, nullptr, nullptr, midb, NN, FFN, HID, 1.0f, 1);
  gemm(midb, f2wb, ffn_b2, hf, y2f, nullptr, NN, HID, FFN, 1.0f, 0);
  layernorm512<<<NN / 8, 256, 0, stream>>>(y2f, ln2_g, ln2_b, (float*)d_out, nullptr, NN);
}
